// VisualAttentionProcess_6743098655329
// MI455X (gfx1250) — compile-verified, matrix operand order checked by reading, not run on hardware
//
#include <hip/hip_runtime.h>

typedef __bf16 bf16_t;
typedef __attribute__((ext_vector_type(16))) __bf16 v16bf;
typedef __attribute__((ext_vector_type(8)))  __bf16 v8bf;
typedef __attribute__((ext_vector_type(8)))  float  v8f;

#define USE_ASYNC_LDS 1

#if USE_ASYNC_LDS
// Copy 16 contiguous bytes global -> LDS via the CDNA5 async data path (ASYNCcnt).
__device__ __forceinline__ void async_copy_b128(void* lds_ptr, const void* gptr) {
  unsigned int lds_off = (unsigned int)(unsigned long long)lds_ptr;  // low 32 bits = LDS offset
  unsigned long long ga = (unsigned long long)gptr;
  asm volatile("global_load_async_to_lds_b128 %0, %1, off"
               :: "v"(lds_off), "v"(ga) : "memory");
}
__device__ __forceinline__ void wait_async0() {
  asm volatile("s_wait_asynccnt 0x0" ::: "memory");
}
#endif

// Build a 16-element bf16 WMMA operand from two contiguous 8-element chunks.
// Per ISA 7.12.2 (16-bit A 16x32): lanes 0-15 hold K {0..7,16..23} of row M=lane,
// lanes 16-31 hold K {8..15,24..31}; B (32x16) is the mirrored layout per column.
__device__ __forceinline__ v16bf pack_ab(const bf16_t* lo, const bf16_t* hi) {
  v8bf a0 = *(const v8bf*)lo;
  v8bf a1 = *(const v8bf*)hi;
  v16bf r;
#pragma unroll
  for (int i = 0; i < 8; ++i) { r[i] = a0[i]; r[i + 8] = a1[i]; }
  return r;
}

// One-time: W[K][N] f32 -> WT[N][K] bf16 (N-major so GEMM B staging is contiguous).
__global__ __launch_bounds__(256) void w_transpose_bf16(
    const float* __restrict__ W, bf16_t* __restrict__ WT, int K, int N) {
  int idx = blockIdx.x * 256 + threadIdx.x;
  if (idx < K * N) {
    int k = idx / N, n = idx - k * N;
    WT[(size_t)n * K + k] = (bf16_t)W[idx];
  }
}

// Tiled GEMM: C[M,N] = A[M,K] @ B[K,N] (*scale + bias), with B given as BT[N][K] bf16.
// Block tile 128x128 (256 threads, 8 waves as 4(M) x 2(N); wave tile 32x64, 8 accums).
// Double-buffered LDS: async copies for step s+1 overlap WMMAs of step s
// (one barrier + one s_wait_asynccnt per K-step).
// Grid: x = N/128 (fast -> A slab stays L2-resident), y = M/128.
// rows_mod/rows_valid: per-batch row remap (zero-padded rows), for K/V projections.
template <bool A_IS_F32, bool OUT_F32>
__global__ __launch_bounds__(256) void gemm_bf16_wmma(
    const void* __restrict__ Av, const bf16_t* __restrict__ BT,
    const float* __restrict__ bias, void* __restrict__ Cv,
    int M, int K, int N, float scale, int rows_mod, int rows_valid) {
  __shared__ bf16_t Als[2][128][32];   // 16 KB ping-pong
  __shared__ bf16_t Bls[2][128][32];   // 16 KB ping-pong, Bls[buf][n][k]

  const int tid  = threadIdx.x;
  const int lane = tid & 31;
  const int w    = tid >> 5;     // 0..7
  const int wm   = w & 3;        // 0..3 : 32-row slab
  const int wn   = w >> 2;       // 0..1 : 64-col slab
  const int r    = lane & 15;
  const int half = lane >> 4;

  const int gn0 = blockIdx.x * 128;
  const int gm0 = blockIdx.y * 128;

  v8f acc[2][4];
#pragma unroll
  for (int ms = 0; ms < 2; ++ms)
#pragma unroll
    for (int nt = 0; nt < 4; ++nt) acc[ms][nt] = v8f{};

  // Staging: 128x32 = 4096 bf16 per tile; 256 threads x 2 vec8 chunks.
  int st_row[2], st_seg[2]; bool a_valid[2]; long a_row[2];
#pragma unroll
  for (int i = 0; i < 2; ++i) {
    int c = tid + i * 256;          // 0..511
    st_row[i] = c >> 2;             // 0..127
    st_seg[i] = (c & 3) * 8;        // 0,8,16,24
    int m_st = gm0 + st_row[i];
    if (rows_mod > 0) {
      int bb = m_st / rows_mod; int l = m_st - bb * rows_mod;
      a_valid[i] = l < rows_valid; a_row[i] = (long)bb * rows_valid + l;
    } else { a_valid[i] = m_st < M; a_row[i] = m_st; }
  }

  auto stage = [&](int buf, int k0) {
#pragma unroll
    for (int i = 0; i < 2; ++i) {
      // ---- A tile ----
      if (a_valid[i]) {
        if constexpr (A_IS_F32) {
          const float* Af = (const float*)Av;
          v8f t = *(const v8f*)(Af + a_row[i] * (long)K + k0 + st_seg[i]);
          v8bf av;
#pragma unroll
          for (int e = 0; e < 8; ++e) av[e] = (bf16_t)t[e];
          *(v8bf*)&Als[buf][st_row[i]][st_seg[i]] = av;
        } else {
          const bf16_t* Ab = (const bf16_t*)Av;
#if USE_ASYNC_LDS
          async_copy_b128(&Als[buf][st_row[i]][st_seg[i]],
                          Ab + a_row[i] * (long)K + k0 + st_seg[i]);
#else
          *(v8bf*)&Als[buf][st_row[i]][st_seg[i]] =
              *(const v8bf*)(Ab + a_row[i] * (long)K + k0 + st_seg[i]);
#endif
        }
      } else {
        v8bf z;
#pragma unroll
        for (int e = 0; e < 8; ++e) z[e] = (bf16_t)0.f;
        *(v8bf*)&Als[buf][st_row[i]][st_seg[i]] = z;
      }
      // ---- B tile (pre-transposed bf16 weights: contiguous 16B copies) ----
      const bf16_t* src = BT + (size_t)(gn0 + st_row[i]) * K + k0 + st_seg[i];
#if USE_ASYNC_LDS
      async_copy_b128(&Bls[buf][st_row[i]][st_seg[i]], src);
#else
      *(v8bf*)&Bls[buf][st_row[i]][st_seg[i]] = *(const v8bf*)src;
#endif
    }
  };

  const int nsteps = K >> 5;
  stage(0, 0);
  for (int s = 0; s < nsteps; ++s) {
#if USE_ASYNC_LDS
    wait_async0();                 // own stage(s) copies have landed
#endif
    __syncthreads();               // everyone's stage(s) landed; compute(s-1) done
    if (s + 1 < nsteps) stage((s + 1) & 1, (s + 1) << 5);  // overlaps compute below
    const int buf = s & 1;

    // ---- WMMA: wave tile 32x64 = 2 a-ops x 4 b-ops -> 8 WMMAs ----
    v16bf a0 = pack_ab(&Als[buf][wm * 32 + r][half * 8],
                       &Als[buf][wm * 32 + r][half * 8 + 16]);
    v16bf a1 = pack_ab(&Als[buf][wm * 32 + 16 + r][half * 8],
                       &Als[buf][wm * 32 + 16 + r][half * 8 + 16]);
#pragma unroll
    for (int nt = 0; nt < 4; ++nt) {
      int cb = wn * 64 + nt * 16 + r;
      v16bf bb = pack_ab(&Bls[buf][cb][half * 8], &Bls[buf][cb][half * 8 + 16]);
      acc[0][nt] = __builtin_amdgcn_wmma_f32_16x16x32_bf16(false, a0, false, bb, (short)0, acc[0][nt], false, false);
      acc[1][nt] = __builtin_amdgcn_wmma_f32_16x16x32_bf16(false, a1, false, bb, (short)0, acc[1][nt], false, false);
    }
  }

  // ---- epilogue: lane holds col N=lane&15; VGPR j holds row j + 8*(lane>>4) ----
#pragma unroll
  for (int ms = 0; ms < 2; ++ms) {
#pragma unroll
    for (int nt = 0; nt < 4; ++nt) {
      int n = gn0 + wn * 64 + nt * 16 + r;
      float bval = bias ? bias[n] : 0.f;
#pragma unroll
      for (int j = 0; j < 8; ++j) {
        int m = gm0 + wm * 32 + ms * 16 + half * 8 + j;
        if (m < M) {
          float val = acc[ms][nt][j] * scale + bval;
          if (OUT_F32) ((float*)Cv)[(size_t)m * N + n] = val;
          else ((bf16_t*)Cv)[(size_t)m * N + n] = (bf16_t)val;
        }
      }
    }
  }
}

// Fused attention core: scores = q@k^T (scale pre-folded into q), softmax(L=77),
// out = probs@v. One workgroup = 4 waves; wave handles a 16-row q tile of one (b,h).
// Kb/Vb are padded to Lp=96 rows per batch (rows 77..95 are zero).
__global__ __launch_bounds__(128) void attn_core(
    const bf16_t* __restrict__ Q, const bf16_t* __restrict__ Kb,
    const bf16_t* __restrict__ Vb, bf16_t* __restrict__ O) {
  __shared__ bf16_t Kls[96][64];      // 12 KB
  __shared__ bf16_t VT[64][96];       // 12 KB (v transposed)
  __shared__ float  Sls[4][16][96];   // 24 KB per-wave scores
  __shared__ bf16_t Pls[4][16][96];   // 12 KB per-wave probs

  const int tid  = threadIdx.x;
  const int lane = tid & 31;
  const int w    = tid >> 5;
  const int r    = lane & 15;
  const int half = lane >> 4;
  const int b    = blockIdx.x >> 4;
  const int h    = blockIdx.x & 15;
  const int s0   = blockIdx.y * 64 + w * 16;

  // ---- stage K (async contiguous copies) and V (transposed, register scatter) ----
#pragma unroll
  for (int j = 0; j < 6; ++j) {
    int e   = tid + j * 128;   // 0..767 vec8 chunks
    int row = e >> 3;          // 0..95
    int seg = (e & 7) * 8;     // 0..56
    size_t goff = ((size_t)(b * 96 + row)) * 1024 + h * 64 + seg;
#if USE_ASYNC_LDS
    async_copy_b128(&Kls[row][seg], Kb + goff);
#else
    *(v8bf*)&Kls[row][seg] = *(const v8bf*)(Kb + goff);
#endif
    v8bf vv = *(const v8bf*)(Vb + goff);
#pragma unroll
    for (int i = 0; i < 8; ++i) VT[seg + i][row] = vv[i];
  }
#if USE_ASYNC_LDS
  wait_async0();
#endif
  __syncthreads();

  // ---- q operands straight from global (bf16, scale already applied) ----
  const bf16_t* qrow = Q + ((size_t)(b * 4096 + s0 + r)) * 1024 + h * 64;
  v16bf qa0 = pack_ab(qrow +      half * 8, qrow +      half * 8 + 16);
  v16bf qa1 = pack_ab(qrow + 32 + half * 8, qrow + 32 + half * 8 + 16);

  // ---- scores: 6 N-tiles x (K=64 -> 2 WMMA) ----
#pragma unroll
  for (int nt = 0; nt < 6; ++nt) {
    v8f acc = {};
    v16bf kb0 = pack_ab(&Kls[nt * 16 + r][     half * 8], &Kls[nt * 16 + r][     half * 8 + 16]);
    v16bf kb1 = pack_ab(&Kls[nt * 16 + r][32 + half * 8], &Kls[nt * 16 + r][32 + half * 8 + 16]);
    acc = __builtin_amdgcn_wmma_f32_16x16x32_bf16(false, qa0, false, kb0, (short)0, acc, false, false);
    acc = __builtin_amdgcn_wmma_f32_16x16x32_bf16(false, qa1, false, kb1, (short)0, acc, false, false);
#pragma unroll
    for (int j = 0; j < 8; ++j) Sls[w][half * 8 + j][nt * 16 + r] = acc[j];
  }
  __syncthreads();

  // ---- softmax over L=77 (one row per lane 0..15 of each wave) ----
  if (lane < 16) {
    float mx = -1e30f;
    for (int j = 0; j < 77; ++j) mx = fmaxf(mx, Sls[w][r][j]);
    float sum = 0.f;
    for (int j = 0; j < 77; ++j) sum += __expf(Sls[w][r][j] - mx);
    float inv = 1.f / sum;
    for (int j = 0; j < 77; ++j) Pls[w][r][j] = (bf16_t)(__expf(Sls[w][r][j] - mx) * inv);
    for (int j = 77; j < 96; ++j) Pls[w][r][j] = (bf16_t)0.f;
  }
  __syncthreads();

  // ---- out = P[16x96] @ V[96x64]: 3 K-steps x 4 N-tiles ----
  v8f oacc[4] = {v8f{}, v8f{}, v8f{}, v8f{}};
#pragma unroll
  for (int ks = 0; ks < 3; ++ks) {
    v16bf pa = pack_ab(&Pls[w][r][ks * 32 + half * 8], &Pls[w][r][ks * 32 + half * 8 + 16]);
#pragma unroll
    for (int nt = 0; nt < 4; ++nt) {
      v16bf vbm = pack_ab(&VT[nt * 16 + r][ks * 32 + half * 8],
                          &VT[nt * 16 + r][ks * 32 + half * 8 + 16]);
      oacc[nt] = __builtin_amdgcn_wmma_f32_16x16x32_bf16(false, pa, false, vbm, (short)0, oacc[nt], false, false);
    }
  }
#pragma unroll
  for (int nt = 0; nt < 4; ++nt) {
#pragma unroll
    for (int j = 0; j < 8; ++j) {
      int s   = s0 + half * 8 + j;
      int col = h * 64 + nt * 16 + r;
      O[((size_t)(b * 4096 + s)) * 1024 + col] = (bf16_t)oacc[nt][j];
    }
  }
}

extern "C" void kernel_launch(void* const* d_in, const int* in_sizes, int n_in,
                              void* d_out, int out_size, void* d_ws, size_t ws_size,
                              hipStream_t stream) {
  (void)in_sizes; (void)n_in; (void)out_size; (void)ws_size;
  const float* hs  = (const float*)d_in[0];  // [16,4096,1024]
  const float* enc = (const float*)d_in[1];  // [16,77,768]
  const float* Wq  = (const float*)d_in[2];  // [1024,1024]
  const float* Wk  = (const float*)d_in[3];  // [768,1024]
  const float* Wv  = (const float*)d_in[4];  // [768,1024]
  const float* Wo  = (const float*)d_in[5];  // [1024,1024]
  const float* bo  = (const float*)d_in[6];  // [1024]
  float* out = (float*)d_out;

  const size_t QBYTES  = (size_t)65536 * 1024 * 2;  // 134 MB
  const size_t KVBYTES = (size_t)1536 * 1024 * 2;   // 3 MB (16 x 96 padded rows)
  const size_t W10BYTES = (size_t)1024 * 1024 * 2;  // 2 MB
  const size_t W76BYTES = (size_t)768 * 1024 * 2;   // 1.5 MB
  char* ws = (char*)d_ws;
  size_t off = 0;
  bf16_t* Qb  = (bf16_t*)(ws + off); off += QBYTES;
  bf16_t* Kp  = (bf16_t*)(ws + off); off += KVBYTES;
  bf16_t* Vp  = (bf16_t*)(ws + off); off += KVBYTES;
  bf16_t* Ab  = (bf16_t*)(ws + off); off += QBYTES;
  bf16_t* WqT = (bf16_t*)(ws + off); off += W10BYTES;
  bf16_t* WkT = (bf16_t*)(ws + off); off += W76BYTES;
  bf16_t* WvT = (bf16_t*)(ws + off); off += W76BYTES;
  bf16_t* WoT = (bf16_t*)(ws + off); off += W10BYTES;

  // One-time weight conversion to bf16, N-major.
  w_transpose_bf16<<<4096, 256, 0, stream>>>(Wq, WqT, 1024, 1024);
  w_transpose_bf16<<<3072, 256, 0, stream>>>(Wk, WkT, 768, 1024);
  w_transpose_bf16<<<3072, 256, 0, stream>>>(Wv, WvT, 768, 1024);
  w_transpose_bf16<<<4096, 256, 0, stream>>>(Wo, WoT, 1024, 1024);

  const float scale = 0.125f;  // 1/sqrt(64), folded into q

  // q = HS @ Wq * scale : grid x = N/128 (fast), y = M/128
  gemm_bf16_wmma<true, false><<<dim3(8, 512), 256, 0, stream>>>(
      hs, WqT, nullptr, Qb, 65536, 1024, 1024, scale, 0, 0);
  // k/v = E @ W (77 -> 96 padded rows per batch)
  gemm_bf16_wmma<true, false><<<dim3(8, 12), 256, 0, stream>>>(
      enc, WkT, nullptr, Kp, 1536, 768, 1024, 1.0f, 96, 77);
  gemm_bf16_wmma<true, false><<<dim3(8, 12), 256, 0, stream>>>(
      enc, WvT, nullptr, Vp, 1536, 768, 1024, 1.0f, 96, 77);
  // fused attention: grid (B*H, S/64)
  attn_core<<<dim3(256, 64), 128, 0, stream>>>(Qb, Kp, Vp, Ab);
  // out = attn @ Wo + bo -> f32
  gemm_bf16_wmma<false, true><<<dim3(8, 512), 256, 0, stream>>>(
      Ab, WoT, bo, out, 65536, 1024, 1024, 1.0f, 0, 0);
}
